// GraphEncoder_25366076850849
// MI455X (gfx1250) — compile-verified, matrix operand order checked by reading, not run on hardware
//
#include <hip/hip_runtime.h>

typedef __attribute__((ext_vector_type(16))) _Float16 v16h;
typedef __attribute__((ext_vector_type(8)))  _Float16 v8h;
typedef __attribute__((ext_vector_type(8)))  float    v8f;

// ---------------------------------------------------------------------------
// helpers
// ---------------------------------------------------------------------------
__device__ __forceinline__ void atomAddF(float* p, float v) {
  __hip_atomic_fetch_add(p, v, __ATOMIC_RELAXED, __HIP_MEMORY_SCOPE_AGENT);
}

__global__ void zero_kernel(float* __restrict__ p, int n) {
  int i = blockIdx.x * blockDim.x + threadIdx.x;
  if (i < n) p[i] = 0.0f;
}

// one-time f32 -> f16 conversion of a weight matrix (tiny, L2-resident after)
__global__ void cvt_f16_kernel(const float* __restrict__ src, _Float16* __restrict__ dst, int n) {
  int i = blockIdx.x * blockDim.x + threadIdx.x;
  if (i < n) dst[i] = (_Float16)src[i];
}

// deg[col] += ew  (self-loop +1 added in dinv_kernel)
__global__ void deg_kernel(const long long* __restrict__ ei, const float* __restrict__ ew,
                           float* __restrict__ deg, int E) {
  int e = blockIdx.x * blockDim.x + threadIdx.x;
  if (e >= E) return;
  int c = (int)ei[(size_t)E + e];
  atomAddF(deg + c, ew[e]);
}

__global__ void dinv_kernel(const float* __restrict__ deg, float* __restrict__ dinv, int n) {
  int i = blockIdx.x * blockDim.x + threadIdx.x;
  if (i < n) dinv[i] = rsqrtf(deg[i] + 1.0f);   // deg >= 1 always (self loop)
}

__global__ void norm_kernel(const long long* __restrict__ ei, const float* __restrict__ ew,
                            const float* __restrict__ dinv, float* __restrict__ nrm, int E) {
  int e = blockIdx.x * blockDim.x + threadIdx.x;
  if (e >= E) return;
  int r = (int)ei[e];
  int c = (int)ei[(size_t)E + e];
  nrm[e] = dinv[r] * ew[e] * dinv[c];
}

// ---------------------------------------------------------------------------
// WMMA GEMM:  H[M, NT*16] = X[M, K] * Wh[NT*16, K]^T
// One wave owns a 16-row M tile and sweeps all NT column tiles, reusing its A
// fragment across NT WMMAs per k-step.  A: f32 loads + in-register cvt
// (amortized over NT WMMAs).  B: pre-converted f16, pure 16B vector loads.
// ---------------------------------------------------------------------------
template<int NT>
__global__ void gemm_wmma(const float* __restrict__ X, const _Float16* __restrict__ Wh,
                          float* __restrict__ H, int Mtiles, int K) {
  const int mt = blockIdx.x * blockDim.y + threadIdx.y;   // one wave per M-tile
  if (mt >= Mtiles) return;                               // wave-uniform exit
  const int lane = threadIdx.x;
  const int r    = lane & 15;      // row within A tile / column within B tile
  const int half = lane >> 4;      // K-half selector per ISA 16-bit layouts

  const float* xrow = X + (size_t)(mt * 16 + r) * K;

  v8f acc[NT];
  const v8f vzero = {0.f, 0.f, 0.f, 0.f, 0.f, 0.f, 0.f, 0.f};
#pragma unroll
  for (int t = 0; t < NT; ++t) acc[t] = vzero;

  for (int kb = 0; kb < K; kb += 32) {
    // A fragment: lane holds row (mt*16+r), K = kb + half*8 + {0..7, 16..23}
    v16h a;
    {
      const float* xp = xrow + kb + half * 8;
      if (kb + 32 < K) __builtin_prefetch(xp + 32, 0, 1);   // next k-step of A
#pragma unroll
      for (int j = 0; j < 8; ++j) {
        a[j]     = (_Float16)xp[j];
        a[8 + j] = (_Float16)xp[16 + j];
      }
    }
#pragma unroll
    for (int t = 0; t < NT; ++t) {
      // B fragment: lane holds column (t*16+r) of B == row (t*16+r) of Wh;
      // two aligned 16B f16 loads, no conversion VALU.
      const _Float16* wp = Wh + (size_t)(t * 16 + r) * K + kb + half * 8;
      v8h blo = *reinterpret_cast<const v8h*>(wp);
      v8h bhi = *reinterpret_cast<const v8h*>(wp + 16);
      v16h b = __builtin_shufflevector(blo, bhi, 0, 1, 2, 3, 4, 5, 6, 7,
                                                 8, 9, 10, 11, 12, 13, 14, 15);
      acc[t] = __builtin_amdgcn_wmma_f32_16x16x32_f16(
          /*neg_a=*/false, a, /*neg_b=*/false, b,
          /*c_mod=*/(short)0, acc[t], /*reuse_a=*/false, /*reuse_b=*/false);
    }
  }

  // C/D layout: vgpr v, lane -> M = v + 8*half, N = lane&15
  const int Nout = NT * 16;
#pragma unroll
  for (int t = 0; t < NT; ++t) {
    float* hp = H + (size_t)(mt * 16 + half * 8) * Nout + t * 16 + r;
#pragma unroll
    for (int v = 0; v < 8; ++v) hp[(size_t)v * Nout] = acc[t][v];
  }
}

// ---------------------------------------------------------------------------
// Edge scatter: acc[col, :] += norm[e] * H[row, :]   (4 channels per thread)
// ---------------------------------------------------------------------------
template<int C>
__global__ void scatter_kernel(const long long* __restrict__ ei, const float* __restrict__ nrm,
                               const float* __restrict__ H, float* __restrict__ acc, int E) {
  constexpr int CH = C / 4;
  int t = blockIdx.x * blockDim.x + threadIdx.x;
  if (t >= E * CH) return;
  int e  = t / CH;
  int cc = (t - e * CH) * 4;
  int r = (int)ei[e];
  int c = (int)ei[(size_t)E + e];
  float nm = nrm[e];
  const float4 h = *reinterpret_cast<const float4*>(H + (size_t)r * C + cc);
  float* dst = acc + (size_t)c * C + cc;
  atomAddF(dst + 0, nm * h.x);
  atomAddF(dst + 1, nm * h.y);
  atomAddF(dst + 2, nm * h.z);
  atomAddF(dst + 3, nm * h.w);
}

// ---------------------------------------------------------------------------
// Finalize: acc += dinv^2 * H (self loop) + bias; then PReLU.  In place.
// ---------------------------------------------------------------------------
template<int C>
__global__ void finalize_kernel(float* __restrict__ acc, const float* __restrict__ H,
                                const float* __restrict__ dinv, const float* __restrict__ b,
                                const float* __restrict__ alpha, int n) {
  int i = blockIdx.x * blockDim.x + threadIdx.x;
  if (i >= n * C) return;
  int node = i / C;
  int c    = i - node * C;
  float di = dinv[node];
  float v  = acc[i] + di * di * H[i] + b[c];
  acc[i] = (v >= 0.0f) ? v : alpha[c] * v;
}

// ---------------------------------------------------------------------------
// launch
// ---------------------------------------------------------------------------
extern "C" void kernel_launch(void* const* d_in, const int* in_sizes, int n_in,
                              void* d_out, int out_size, void* d_ws, size_t ws_size,
                              hipStream_t stream) {
  constexpr int IN_C = 512, HID = 128, HID2 = 64;

  const float*     x      = (const float*)d_in[0];
  const long long* ei     = (const long long*)d_in[1];   // int64 edge_index [2, E]
  const float*     ew     = (const float*)d_in[2];
  const float*     W1     = (const float*)d_in[3];
  const float*     b1     = (const float*)d_in[4];
  const float*     alpha1 = (const float*)d_in[5];
  const float*     W2     = (const float*)d_in[6];
  const float*     b2     = (const float*)d_in[7];
  const float*     alpha2 = (const float*)d_in[8];
  float*           out    = (float*)d_out;

  const int Nn = in_sizes[0] / IN_C;   // 50000
  const int E  = in_sizes[2];          // 800000
  const int Mtiles = Nn / 16;          // 3125 (exact)

  // workspace layout (float units; all vector-load bases stay 16B aligned)
  float* ws   = (float*)d_ws;
  float* deg  = ws;
  float* dinv = deg  + Nn;
  float* nrm  = dinv + Nn;
  float* H1   = nrm  + E;
  float* Z1   = H1   + (size_t)Nn * HID;
  float* H2   = Z1   + (size_t)Nn * HID;
  _Float16* W1h = (_Float16*)(H2 + (size_t)Nn * HID2);
  _Float16* W2h = W1h + (size_t)HID * IN_C;

  const int TPB = 256;
  auto blocks = [](long long n, int tpb) { return (unsigned)((n + tpb - 1) / tpb); };

  // --- one-time weight conversion to f16 (tiny; stays hot in L2) ---
  cvt_f16_kernel<<<blocks(HID * IN_C, TPB), TPB, 0, stream>>>(W1, W1h, HID * IN_C);
  cvt_f16_kernel<<<blocks(HID2 * HID, TPB), TPB, 0, stream>>>(W2, W2h, HID2 * HID);

  // --- normalization factors (shared by both layers) ---
  zero_kernel<<<blocks(Nn, TPB), TPB, 0, stream>>>(deg, Nn);
  deg_kernel<<<blocks(E, TPB), TPB, 0, stream>>>(ei, ew, deg, E);
  dinv_kernel<<<blocks(Nn, TPB), TPB, 0, stream>>>(deg, dinv, Nn);
  norm_kernel<<<blocks(E, TPB), TPB, 0, stream>>>(ei, ew, dinv, nrm, E);

  // --- layer 1 ---
  {
    dim3 blk(32, 4);
    dim3 grd((Mtiles + 3) / 4);
    gemm_wmma<HID / 16><<<grd, blk, 0, stream>>>(x, W1h, H1, Mtiles, IN_C);
  }
  zero_kernel<<<blocks((long long)Nn * HID, TPB), TPB, 0, stream>>>(Z1, Nn * HID);
  scatter_kernel<HID><<<blocks((long long)E * (HID / 4), TPB), TPB, 0, stream>>>(ei, nrm, H1, Z1, E);
  finalize_kernel<HID><<<blocks((long long)Nn * HID, TPB), TPB, 0, stream>>>(Z1, H1, dinv, b1, alpha1, Nn);

  // --- layer 2 ---
  {
    dim3 blk(32, 4);
    dim3 grd((Mtiles + 3) / 4);
    gemm_wmma<HID2 / 16><<<grd, blk, 0, stream>>>(Z1, W2h, H2, Mtiles, HID);
  }
  zero_kernel<<<blocks((long long)Nn * HID2, TPB), TPB, 0, stream>>>(out, Nn * HID2);
  scatter_kernel<HID2><<<blocks((long long)E * (HID2 / 4), TPB), TPB, 0, stream>>>(ei, nrm, H2, out, E);
  finalize_kernel<HID2><<<blocks((long long)Nn * HID2, TPB), TPB, 0, stream>>>(out, H2, dinv, b2, alpha2, Nn);
}
